// SimpleAttentionSemanticEvolution_20151986553033
// MI455X (gfx1250) — matrix operand order repaired and checked by reading, not compiled, not run on hardware
//
#include <hip/hip_runtime.h>
#include <hip/hip_bf16.h>

#define BATCH 4
#define CCH   512
#define HW    4096
#define NTXT  256
#define TDIM  1024
#define HEADS 8
#define DHEAD 128
#define KVSPLIT 4
#define KVSEG (HW / KVSPLIT)    // 1024 kv positions per wave

typedef __attribute__((ext_vector_type(16))) _Float16 v16h;
typedef __attribute__((ext_vector_type(8)))  _Float16 v8h;
typedef __attribute__((ext_vector_type(8)))  float    v8f;

__device__ inline v8f wmma_f16(v16h a, v16h b, v8f c) {
    return __builtin_amdgcn_wmma_f32_16x16x32_f16(false, a, false, b, (short)0, c, false, false);
}

// ---------------- conversion kernels ----------------

// LDS-tiled transpose + f32->f16 (+scale): dst[b][c][r] = (f16)(scale * src[b][r][c])
// grid = (cols/64, rows/64, batch), block = 256. rows%64==0, cols%64==0.
__global__ __launch_bounds__(256) void cvt_transpose_tiled(const float* __restrict__ src,
                                                           _Float16* __restrict__ dst,
                                                           int rows, int cols, float scale) {
    __shared__ _Float16 tile[64][65];
    int b  = blockIdx.z;
    int r0 = blockIdx.y << 6;
    int c0 = blockIdx.x << 6;
    const float* S = src + ((size_t)b * rows + r0) * cols + c0;
    _Float16*    D = dst + ((size_t)b * cols + c0) * rows + r0;
    int tid = threadIdx.x;
#pragma unroll
    for (int i = 0; i < 16; ++i) {               // coalesced f32 reads
        int lin = tid + i * 256;
        int rr = lin >> 6, cc = lin & 63;
        tile[rr][cc] = (_Float16)(S[(size_t)rr * cols + cc] * scale);
    }
    __syncthreads();
#pragma unroll
    for (int i = 0; i < 16; ++i) {               // contiguous f16 writes
        int lin = tid + i * 256;
        int rr = lin >> 6, cc = lin & 63;
        D[(size_t)rr * rows + cc] = tile[cc][rr];
    }
}

__global__ void cvt_f32_f16(const float* __restrict__ src, _Float16* __restrict__ dst, int n) {
    int i = blockIdx.x * blockDim.x + threadIdx.x;
    if (i < n) dst[i] = (_Float16)src[i];
}

// ---------------- 64x64-tile WMMA GEMM ----------------
// A: [M,K] f16 row-major.  Bt: [N,K] f16 (B transposed).  K%32==0, M%64==0, N%64==0.
// MODE 0: C f16 row-major [M,N]
// MODE 1: C f16 transposed per batch: b=m/rowsPB, j=m%rowsPB -> C[(b*N+n)*rowsPB + j]
// MODE 2: C f32 row-major [M,N] + bias[n]
template<int MODE>
__global__ __launch_bounds__(128) void gemm64(const _Float16* __restrict__ A,
                                              const _Float16* __restrict__ Bt,
                                              void* __restrict__ Cout,
                                              const float* __restrict__ bias,
                                              int M, int N, int K, int rowsPB) {
    int wave = (blockIdx.x << 2) + (threadIdx.x >> 5);
    int lane = threadIdx.x & 31;
    int lh = lane & 15;
    int hi = lane >> 4;
    int tilesN = N >> 6;
    int mbase = (wave / tilesN) << 6;
    int nbase = (wave % tilesN) << 6;
    if (mbase >= M) return;

    const _Float16* Arow[4];
#pragma unroll
    for (int i = 0; i < 4; ++i)
        Arow[i] = A + (size_t)(mbase + i * 16 + lh) * K + (hi ? 16 : 0);

    v8f acc[4][4] = {};

    for (int kk = 0; kk < K; kk += 32) {
        v16h a[4];
#pragma unroll
        for (int i = 0; i < 4; ++i) {
            const _Float16* ap = Arow[i] + kk;
            *(v8h*)&a[i]         = *(const v8h*)ap;
            *(((v8h*)&a[i]) + 1) = *(const v8h*)(ap + 8);
        }
#pragma unroll
        for (int t = 0; t < 4; ++t) {
            int n = nbase + t * 16 + lh;
            v16h bf = *(const v16h*)(Bt + (size_t)n * K + kk + (hi ? 16 : 0));
#pragma unroll
            for (int i = 0; i < 4; ++i)
                acc[i][t] = wmma_f16(a[i], bf, acc[i][t]);
        }
    }

    int rowoff = hi ? 8 : 0;
#pragma unroll
    for (int i = 0; i < 4; ++i) {
#pragma unroll
        for (int t = 0; t < 4; ++t) {
#pragma unroll
            for (int r = 0; r < 8; ++r) {
                int m = mbase + i * 16 + rowoff + r;
                int n = nbase + t * 16 + lh;
                float v = acc[i][t][r];
                if (MODE == 0) {
                    ((_Float16*)Cout)[(size_t)m * N + n] = (_Float16)v;
                } else if (MODE == 1) {
                    int b = m / rowsPB;
                    int j = m % rowsPB;
                    ((_Float16*)Cout)[((size_t)b * N + n) * rowsPB + j] = (_Float16)v;
                } else {
                    ((float*)Cout)[(size_t)m * N + n] = v + bias[n];
                }
            }
        }
    }
}

// ---------------- fused flash attention, KV-split partials ----------------
// 4 waves/block; wave w handles KV segment [w*1024, (w+1)*1024) for one
// (b, head, 16-query) tile. Q is pre-scaled by 1/sqrt(DHEAD) at conversion.
// Row-sum of P computed with WMMA against an all-ones B fragment (no shuffles);
// O/l rescale skipped (wave-uniform branch) when the running max is unchanged.
__global__ __launch_bounds__(128) void attn_flash(const _Float16* __restrict__ Q,
                                                  const _Float16* __restrict__ Kmat,
                                                  const _Float16* __restrict__ Vt,
                                                  float* __restrict__ Opart,
                                                  float* __restrict__ Mpart,
                                                  float* __restrict__ Lpart) {
    __shared__ _Float16 Pall[4][16 * 64];
    int wid  = threadIdx.x >> 5;          // KV split id
    int lane = threadIdx.x & 31;
    int lh = lane & 15;
    int hi = lane >> 4;
    int tile = blockIdx.x;                // b*128 + h*16 + qt
    int qt = tile & 15;
    int h  = (tile >> 4) & 7;
    int b  = tile >> 7;
    int q0 = qt << 4;
    int j_begin = wid * KVSEG;
    _Float16* P = Pall[wid];

    // Q A-fragments for DHEAD=128 (4 K-chunks of 32)
    const _Float16* Qrow = Q + ((size_t)(b * NTXT + q0 + lh)) * TDIM + h * DHEAD;
    v16h qa[4];
#pragma unroll
    for (int kk = 0; kk < 4; ++kk) {
        const _Float16* qp = Qrow + kk * 32 + (hi ? 16 : 0);
        *(v8h*)&qa[kk]         = *(const v8h*)qp;
        *(((v8h*)&qa[kk]) + 1) = *(const v8h*)(qp + 8);
    }

    v16h ones;
#pragma unroll
    for (int i = 0; i < 16; ++i) ones[i] = (_Float16)1.0f;

    v8f o[8] = {};
    v8f lacc = {};
    float mrow[8], corr[8];
#pragma unroll
    for (int r = 0; r < 8; ++r) mrow[r] = -1e30f;

    for (int j0 = j_begin; j0 < j_begin + KVSEG; j0 += 64) {
        // S = Q*K^T for four 16x16 n-tiles (64 kv positions)
        const _Float16* kp = Kmat + ((size_t)(b * HW + j0 + lh)) * TDIM + h * DHEAD + hi * 16;
        v8f s[4] = {};
#pragma unroll
        for (int u = 0; u < 4; ++u) {
            const _Float16* kpu = kp + (size_t)(u * 16) * TDIM;
#pragma unroll
            for (int kk = 0; kk < 4; ++kk) {
                v16h bfrag = *(const v16h*)(kpu + kk * 32);
                s[u] = wmma_f16(qa[kk], bfrag, s[u]);
            }
        }
        __builtin_prefetch(kp + (size_t)64 * TDIM, 0, 1);   // next K chunk

        // online softmax: row max via half-wave xor-reduction only
        int newmax = 0;
#pragma unroll
        for (int r = 0; r < 8; ++r) {
            float v0 = s[0][r], v1 = s[1][r], v2 = s[2][r], v3 = s[3][r];
            float cm = fmaxf(fmaxf(v0, v1), fmaxf(v2, v3));
            cm = fmaxf(cm, __shfl_xor(cm, 1, 32));
            cm = fmaxf(cm, __shfl_xor(cm, 2, 32));
            cm = fmaxf(cm, __shfl_xor(cm, 4, 32));
            cm = fmaxf(cm, __shfl_xor(cm, 8, 32));
            float mold = mrow[r];
            float mn = fmaxf(mold, cm);
            if (mn > mold) newmax = 1;
            corr[r] = __expf(mold - mn);
            mrow[r] = mn;
            float p0 = __expf(v0 - mn), p1 = __expf(v1 - mn);
            float p2 = __expf(v2 - mn), p3 = __expf(v3 - mn);
            int row = hi ? (r + 8) : r;
            P[row * 64 + lh]      = (_Float16)p0;
            P[row * 64 + 16 + lh] = (_Float16)p1;
            P[row * 64 + 32 + lh] = (_Float16)p2;
            P[row * 64 + 48 + lh] = (_Float16)p3;
        }
        if (__ballot(newmax)) {                 // wave-uniform: rescale only if max moved
#pragma unroll
            for (int t = 0; t < 8; ++t)
#pragma unroll
                for (int r = 0; r < 8; ++r) o[t][r] *= corr[r];
#pragma unroll
            for (int r = 0; r < 8; ++r) lacc[r] *= corr[r];
        }

        // reload P (16x64) as two A-fragments from LDS
        v16h pa0, pa1;
        const _Float16* pp = P + lh * 64 + (hi ? 16 : 0);
        *(v8h*)&pa0         = *(const v8h*)pp;
        *(((v8h*)&pa0) + 1) = *(const v8h*)(pp + 8);
        *(v8h*)&pa1         = *(const v8h*)(pp + 32);
        *(((v8h*)&pa1) + 1) = *(const v8h*)(pp + 40);

        // running denominator l += P x ones (row sum via WMMA, no shuffles)
        lacc = wmma_f16(pa0, ones, lacc);
        lacc = wmma_f16(pa1, ones, lacc);

        // O += P x V : 8 n-tiles over DHEAD, contiguous B-frags from Vt
#pragma unroll
        for (int t = 0; t < 8; ++t) {
            const _Float16* vp = Vt + ((size_t)(b * TDIM + h * DHEAD + t * 16 + lh)) * HW
                               + j0 + hi * 16;
            v16h vb0 = *(const v16h*)vp;
            v16h vb1 = *(const v16h*)(vp + 32);
            o[t] = wmma_f16(pa0, vb0, o[t]);
            o[t] = wmma_f16(pa1, vb1, o[t]);
        }
    }

    // store un-normalized partials
    float* Ob = Opart + ((size_t)tile * KVSPLIT + wid) * 16 * DHEAD;
#pragma unroll
    for (int t = 0; t < 8; ++t) {
#pragma unroll
        for (int r = 0; r < 8; ++r) {
            int row = hi ? (r + 8) : r;
            Ob[row * DHEAD + t * 16 + lh] = o[t][r];
        }
    }
    if (lh == 0) {
        size_t base = ((size_t)tile * KVSPLIT + wid) * 16;
#pragma unroll
        for (int r = 0; r < 8; ++r) {
            int row = hi ? (r + 8) : r;
            Mpart[base + row] = mrow[r];
            Lpart[base + row] = lacc[r];
        }
    }
}

// ---------------- flash-decoding combine over KV splits ----------------
__global__ __launch_bounds__(128) void attn_combine(const float* __restrict__ Opart,
                                                    const float* __restrict__ Mpart,
                                                    const float* __restrict__ Lpart,
                                                    _Float16* __restrict__ A16) {
    int tile = blockIdx.x;
    int col  = threadIdx.x;               // 0..127
    int qt = tile & 15;
    int h  = (tile >> 4) & 7;
    int b  = tile >> 7;
    const float* Ob = Opart + (size_t)tile * KVSPLIT * 16 * DHEAD;
    const float* Mb = Mpart + (size_t)tile * KVSPLIT * 16;
    const float* Lb = Lpart + (size_t)tile * KVSPLIT * 16;
    for (int row = 0; row < 16; ++row) {
        float m0 = Mb[0 * 16 + row], m1 = Mb[1 * 16 + row];
        float m2 = Mb[2 * 16 + row], m3 = Mb[3 * 16 + row];
        float mm = fmaxf(fmaxf(m0, m1), fmaxf(m2, m3));
        float w0 = __expf(m0 - mm), w1 = __expf(m1 - mm);
        float w2 = __expf(m2 - mm), w3 = __expf(m3 - mm);
        float l = w0 * Lb[0 * 16 + row] + w1 * Lb[1 * 16 + row]
                + w2 * Lb[2 * 16 + row] + w3 * Lb[3 * 16 + row];
        float acc = w0 * Ob[(0 * 16 + row) * DHEAD + col]
                  + w1 * Ob[(1 * 16 + row) * DHEAD + col]
                  + w2 * Ob[(2 * 16 + row) * DHEAD + col]
                  + w3 * Ob[(3 * 16 + row) * DHEAD + col];
        A16[((size_t)(b * NTXT) + qt * 16 + row) * TDIM + h * DHEAD + col] =
            (_Float16)(acc / l);
    }
}

// ---------------- launcher ----------------

extern "C" void kernel_launch(void* const* d_in, const int* in_sizes, int n_in,
                              void* d_out, int out_size, void* d_ws, size_t ws_size,
                              hipStream_t stream) {
    const float* x   = (const float*)d_in[0];
    // d_in[1] = timesteps (unused by the reference output)
    const float* ctx = (const float*)d_in[2];
    const float* Wq  = (const float*)d_in[3];
    const float* Wk  = (const float*)d_in[4];
    const float* Wv  = (const float*)d_in[5];
    const float* Wo  = (const float*)d_in[6];
    const float* bo  = (const float*)d_in[7];
    float* out = (float*)d_out;

    char* ws = (char*)d_ws;
    size_t off = 0;
    auto carveH = [&](size_t n) -> _Float16* {
        _Float16* p = (_Float16*)(ws + off);
        off = (off + n * sizeof(_Float16) + 255) & ~(size_t)255;
        return p;
    };
    auto carveF = [&](size_t n) -> float* {
        float* p = (float*)(ws + off);
        off = (off + n * sizeof(float) + 255) & ~(size_t)255;
        return p;
    };

    _Float16* vis16 = carveH((size_t)BATCH * HW * CCH);     // [B, HW, C]
    _Float16* ctx16 = carveH((size_t)BATCH * NTXT * TDIM);
    _Float16* Wqt   = carveH((size_t)TDIM * TDIM);          // [N, K], pre-scaled
    _Float16* Wkt   = carveH((size_t)TDIM * CCH);
    _Float16* Wvt   = carveH((size_t)TDIM * CCH);
    _Float16* Wot   = carveH((size_t)TDIM * TDIM);
    _Float16* Q16   = carveH((size_t)BATCH * NTXT * TDIM);
    _Float16* K16   = carveH((size_t)BATCH * HW * TDIM);
    _Float16* Vt16  = carveH((size_t)BATCH * TDIM * HW);    // [B, TDIM, HW]
    _Float16* A16   = carveH((size_t)BATCH * NTXT * TDIM);
    int ntiles = BATCH * HEADS * (NTXT / 16);               // 512
    float* Opart = carveF((size_t)ntiles * KVSPLIT * 16 * DHEAD);
    float* Mpart = carveF((size_t)ntiles * KVSPLIT * 16);
    float* Lpart = carveF((size_t)ntiles * KVSPLIT * 16);
    (void)ws_size; (void)in_sizes; (void)n_in; (void)out_size;

    const float qscale = 0.08838834764831845f;              // 1/sqrt(DHEAD)

    // --- f32 -> f16 staging (LDS-tiled transposes: coalesced both directions) ---
    cvt_transpose_tiled<<<dim3(HW / 64, CCH / 64, BATCH), 256, 0, stream>>>(x, vis16, CCH, HW, 1.0f);
    {
        int n = BATCH * NTXT * TDIM;
        cvt_f32_f16<<<(n + 255) / 256, 256, 0, stream>>>(ctx, ctx16, n);
    }
    cvt_transpose_tiled<<<dim3(TDIM / 64, TDIM / 64, 1), 256, 0, stream>>>(Wq, Wqt, TDIM, TDIM, qscale);
    cvt_transpose_tiled<<<dim3(TDIM / 64, CCH  / 64, 1), 256, 0, stream>>>(Wk, Wkt, CCH,  TDIM, 1.0f);
    cvt_transpose_tiled<<<dim3(TDIM / 64, CCH  / 64, 1), 256, 0, stream>>>(Wv, Wvt, CCH,  TDIM, 1.0f);
    cvt_transpose_tiled<<<dim3(TDIM / 64, TDIM / 64, 1), 256, 0, stream>>>(Wo, Wot, TDIM, TDIM, 1.0f);

    // --- projections (64x64 wave tiles) ---
    {   // Q = ctx16 @ (Wq * qscale)
        int M = BATCH * NTXT, N = TDIM, K = TDIM;
        int waves = (M / 64) * (N / 64);
        gemm64<0><<<waves / 4, 128, 0, stream>>>(ctx16, Wqt, Q16, nullptr, M, N, K, 1);
    }
    {   // K = vis16 @ Wk
        int M = BATCH * HW, N = TDIM, K = CCH;
        int waves = (M / 64) * (N / 64);
        gemm64<0><<<waves / 4, 128, 0, stream>>>(vis16, Wkt, K16, nullptr, M, N, K, 1);
    }
    {   // V = vis16 @ Wv, stored transposed per batch -> Vt16 [B, TDIM, HW]
        int M = BATCH * HW, N = TDIM, K = CCH;
        int waves = (M / 64) * (N / 64);
        gemm64<1><<<waves / 4, 128, 0, stream>>>(vis16, Wvt, Vt16, nullptr, M, N, K, HW);
    }

    // --- fused attention: KV-split flash + combine ---
    attn_flash<<<ntiles, 128, 0, stream>>>(Q16, K16, Vt16, Opart, Mpart, Lpart);
    attn_combine<<<ntiles, 128, 0, stream>>>(Opart, Mpart, Lpart, A16);

    // --- final projection with bias, f32 output ---
    {
        int M = BATCH * NTXT, N = TDIM, K = TDIM;
        int waves = (M / 64) * (N / 64);
        gemm64<2><<<waves / 4, 128, 0, stream>>>(A16, Wot, out, bo, M, N, K, 1);
    }
}
